// GATRegression_21165598834731
// MI455X (gfx1250) — matrix operand order repaired and checked by reading, not compiled, not run on hardware
//
#include <hip/hip_runtime.h>
#include <hip/hip_bf16.h>

// ---------------------------------------------------------------------------
// GATRegression for MI455X (gfx1250, wave32).
//   N=50000 nodes, E=1.6M edges (+N self loops), F=128 = H(4) x C(32), G=64.
// Dense GEMMs use v_wmma_f32_16x16x32_f16 (f32 accumulate); everything else
// is L2-resident atomic scatter/gather (dominant cost ~845MB reads + 211M
// f32 atomics per layer -> atomics resolve in the 192MB L2).
// N % 16 == 0, so the GEMM has no row remainder: stores are unguarded and
// fully coalesced. Next-tile prefetch overlaps HBM fetch with WMMA issue.
// ---------------------------------------------------------------------------

typedef __attribute__((ext_vector_type(16))) _Float16 v16h;
typedef __attribute__((ext_vector_type(8)))  float    v8f;

#define FDIM 128
#define NHEAD 4
#define CHEAD 32

// ---------------------------------------------------------------------------
// GEMM: Y[N,128] = X[N,128] @ W[128,128]   (one wave per 16-row tile)
// A layout (16x32 f16): lane l -> row = l&15; element j -> K = kt*32 +
//   (j>>3)*16 + (l>>4)*8 + (j&7).
// B layout (32x16 f16): lane l -> col = l&15; element j -> K = kt*32 +
//   (j>>3)*16 + (l>>4)*8 + (j&7).
// D layout (16x16 f32): lane l, vgpr r -> row = r + 8*(l>>4), col = l&15.
// ---------------------------------------------------------------------------
__global__ __launch_bounds__(32) void gat_gemm_wmma(
    const float* __restrict__ X, const float* __restrict__ W,
    float* __restrict__ Y, int nrows)
{
    const int lane = threadIdx.x & 31;
    const int row0 = blockIdx.x * 16;
    const int m    = lane & 15;
    const int hi   = lane >> 4;

    // prefetch the A rows of the next 16-row tile (global_prefetch_b8)
    if (row0 + 16 < nrows)
        __builtin_prefetch(X + (long)(row0 + 16 + m) * FDIM, 0, 3);

    v16h a[4];
    const float* xr = X + (long)(row0 + m) * FDIM;
#pragma unroll
    for (int kt = 0; kt < 4; ++kt) {
#pragma unroll
        for (int j = 0; j < 16; ++j) {
            const int k = kt * 32 + (j >> 3) * 16 + hi * 8 + (j & 7);
            a[kt][j] = (_Float16)xr[k];
        }
    }

    float* yr = Y + (long)row0 * FDIM;
#pragma unroll
    for (int nt = 0; nt < 8; ++nt) {
        v8f c = {0.f, 0.f, 0.f, 0.f, 0.f, 0.f, 0.f, 0.f};
        const int col = nt * 16 + m;
#pragma unroll
        for (int kt = 0; kt < 4; ++kt) {
            v16h b;
            const int kb = kt * 32 + hi * 8;
#pragma unroll
            for (int j = 0; j < 16; ++j)
                b[j] = (_Float16)W[(kb + (j >> 3) * 16 + (j & 7)) * FDIM + col];
            c = __builtin_amdgcn_wmma_f32_16x16x32_f16(
                    false, a[kt], false, b, (short)0, c, false, false);
        }
        // N is a multiple of 16: unguarded, coalesced stores
#pragma unroll
        for (int r = 0; r < 8; ++r)
            yr[(long)(r + 8 * hi) * FDIM + col] = c[r];
    }
}

// ---------------------------------------------------------------------------
// Per-(node, head) attention logits: alpha_s/d[n,h] = sum_c xw[n,h,c]*a[h,c]
// ---------------------------------------------------------------------------
__global__ void gat_alpha(const float* __restrict__ xw,
                          const float* __restrict__ a_src,
                          const float* __restrict__ a_dst,
                          float* __restrict__ as_, float* __restrict__ ad_,
                          int n)
{
    const int i = blockIdx.x * blockDim.x + threadIdx.x;
    if (i >= n * NHEAD) return;
    const int node = i >> 2, h = i & 3;
    const float* p = xw + (long)node * FDIM + h * CHEAD;
    float s = 0.f, d = 0.f;
#pragma unroll
    for (int c = 0; c < CHEAD; ++c) {
        s += p[c] * a_src[h * CHEAD + c];
        d += p[c] * a_dst[h * CHEAD + c];
    }
    as_[i] = s;
    ad_[i] = d;
}

__global__ void fill_f32(float* __restrict__ p, float v, long n)
{
    const long i = blockIdx.x * (long)blockDim.x + threadIdx.x;
    if (i < n) p[i] = v;
}

__device__ __forceinline__ void edge_nodes(const long long* __restrict__ ei,
                                           int E_main, int eid,
                                           int& s, int& d)
{
    if (eid < E_main) {
        s = (int)ei[eid];
        d = (int)ei[(long)E_main + eid];
    } else {
        s = d = eid - E_main;   // appended self loops
    }
}

// float atomic max via signed/unsigned int ordering trick (init -inf)
__device__ __forceinline__ void atomicMaxF(float* addr, float v)
{
    if (v >= 0.f) atomicMax((int*)addr, __float_as_int(v));
    else          atomicMin((unsigned int*)addr, (unsigned int)__float_as_int(v));
}

// e = leaky_relu(as[src]+ad[dst]); segment max over dst
__global__ void gat_edge_emax(const long long* __restrict__ ei, int E_main, int Etot,
                              const float* __restrict__ as_, const float* __restrict__ ad_,
                              float* __restrict__ ebuf, float* __restrict__ mmax)
{
    const long i = blockIdx.x * (long)blockDim.x + threadIdx.x;
    if (i >= (long)Etot * NHEAD) return;
    const int eid = (int)(i >> 2), h = (int)(i & 3);
    int s, d;
    edge_nodes(ei, E_main, eid, s, d);
    float v = as_[s * NHEAD + h] + ad_[d * NHEAD + h];
    v = v > 0.f ? v : 0.2f * v;
    ebuf[i] = v;
    atomicMaxF(&mmax[d * NHEAD + h], v);
}

// ex = exp(e - m[dst]); segment sum over dst
__global__ void gat_edge_exp(const long long* __restrict__ ei, int E_main, int Etot,
                             const float* __restrict__ mmax,
                             float* __restrict__ ebuf, float* __restrict__ den)
{
    const long i = blockIdx.x * (long)blockDim.x + threadIdx.x;
    if (i >= (long)Etot * NHEAD) return;
    const int eid = (int)(i >> 2), h = (int)(i & 3);
    int s, d;
    edge_nodes(ei, E_main, eid, s, d);
    const float ex = __expf(ebuf[i] - mmax[d * NHEAD + h]);
    ebuf[i] = ex;
    atomicAdd(&den[d * NHEAD + h], ex);
}

// wave per edge: acc[dst] += xw[src] * (ex/den[dst])  (lane -> 4 channels)
__global__ __launch_bounds__(256) void gat_scatter(
    const long long* __restrict__ ei, int E_main, int Etot,
    const float* __restrict__ xw, const float* __restrict__ ebuf,
    const float* __restrict__ den, float* __restrict__ acc)
{
    const int wid  = (int)((blockIdx.x * (long)blockDim.x + threadIdx.x) >> 5);
    const int lane = threadIdx.x & 31;
    const int nw   = (int)((gridDim.x * (long)blockDim.x) >> 5);
    const int h    = lane >> 3;              // 4 channels/lane stay in one head
    for (int eid = wid; eid < Etot; eid += nw) {
        int s, d;
        edge_nodes(ei, E_main, eid, s, d);
        const float alpha = ebuf[(long)eid * NHEAD + h] / den[d * NHEAD + h];
        const float4 v = *(const float4*)(xw + (long)s * FDIM + lane * 4);
        float* o = acc + (long)d * FDIM + lane * 4;
        atomicAdd(o + 0, v.x * alpha);
        atomicAdd(o + 1, v.y * alpha);
        atomicAdd(o + 2, v.z * alpha);
        atomicAdd(o + 3, v.w * alpha);
    }
}

// h = elu(acc + bias)
__global__ void gat_bias_elu(const float* __restrict__ acc,
                             const float* __restrict__ bias,
                             float* __restrict__ out, long n)
{
    const long i = blockIdx.x * (long)blockDim.x + threadIdx.x;
    if (i >= n) return;
    const float v = acc[i] + bias[(int)(i & (FDIM - 1))];
    out[i] = v > 0.f ? v : (__expf(v) - 1.f);
}

// pooled[batch[n]] += h[n];  cnt[batch[n]] += 1
__global__ void pool_accum(const float* __restrict__ hbuf,
                           const long long* __restrict__ batch,
                           float* __restrict__ pooled, float* __restrict__ cnt,
                           int n)
{
    const int idx = blockIdx.x * blockDim.x + threadIdx.x;
    const int node = idx >> 5, lane = idx & 31;
    if (node >= n) return;
    const int g = (int)batch[node];
    const float4 v = *(const float4*)(hbuf + (long)node * FDIM + lane * 4);
    float* o = pooled + g * FDIM + lane * 4;
    atomicAdd(o + 0, v.x);
    atomicAdd(o + 1, v.y);
    atomicAdd(o + 2, v.z);
    atomicAdd(o + 3, v.w);
    if (lane == 0) atomicAdd(&cnt[g], 1.0f);
}

// out[g] = (pooled[g]/max(cnt,1)) . lin_w + lin_b   (one wave per graph)
__global__ __launch_bounds__(32) void final_linear(
    const float* __restrict__ pooled, const float* __restrict__ cnt,
    const float* __restrict__ lw, const float* __restrict__ lb,
    float* __restrict__ out)
{
    const int g = blockIdx.x, lane = threadIdx.x;
    const float c = fmaxf(cnt[g], 1.0f);
    float s = 0.f;
#pragma unroll
    for (int j = 0; j < 4; ++j) {
        const int k = lane * 4 + j;
        s += (pooled[g * FDIM + k] / c) * lw[k];
    }
#pragma unroll
    for (int off = 16; off > 0; off >>= 1) s += __shfl_down(s, off, 32);
    if (lane == 0) out[g] = s + lb[0];
}

// ---------------------------------------------------------------------------
extern "C" void kernel_launch(void* const* d_in, const int* in_sizes, int n_in,
                              void* d_out, int out_size, void* d_ws, size_t ws_size,
                              hipStream_t stream)
{
    (void)in_sizes; (void)n_in; (void)out_size; (void)ws_size;

    constexpr int N = 50000, E = 1600000, G = 64;
    constexpr int Etot = E + N;

    const float*     x      = (const float*)d_in[0];
    const long long* ei     = (const long long*)d_in[1];   // [2,E] int64
    const long long* batch  = (const long long*)d_in[2];   // [N]   int64
    const float*     W0     = (const float*)d_in[3];
    const float*     asrc0  = (const float*)d_in[4];
    const float*     adst0  = (const float*)d_in[5];
    const float*     b0     = (const float*)d_in[6];
    const float*     W1     = (const float*)d_in[7];
    const float*     asrc1  = (const float*)d_in[8];
    const float*     adst1  = (const float*)d_in[9];
    const float*     b1     = (const float*)d_in[10];
    const float*     lin_w  = (const float*)d_in[11];
    const float*     lin_b  = (const float*)d_in[12];
    float*           outp   = (float*)d_out;

    // workspace carve-up (~106 MB of f32)
    float* xw     = (float*)d_ws;                     // N*F
    float* acc    = xw     + (size_t)N * FDIM;        // N*F
    float* hbuf   = acc    + (size_t)N * FDIM;        // N*F
    float* as_    = hbuf   + (size_t)N * FDIM;        // N*H
    float* ad_    = as_    + (size_t)N * NHEAD;       // N*H
    float* mmax   = ad_    + (size_t)N * NHEAD;       // N*H
    float* den    = mmax   + (size_t)N * NHEAD;       // N*H
    float* pooled = den    + (size_t)N * NHEAD;       // G*F
    float* cnt    = pooled + (size_t)G * FDIM;        // G
    float* ebuf   = cnt    + G;                       // Etot*H

    const long nNF = (long)N * FDIM;
    const long nNH = (long)N * NHEAD;
    const long nEH = (long)Etot * NHEAD;

    const int gGemm  = N / 16;                               // 3125 exact
    const int gAlpha = (int)((nNH + 255) / 256);
    const int gNF    = (int)((nNF + 255) / 256);
    const int gNH    = (int)((nNH + 255) / 256);
    const int gEH    = (int)((nEH + 255) / 256);
    const int gScat  = 4096;                                 // grid-stride waves
    const int gPool  = (N * 32 + 255) / 256;

    for (int layer = 0; layer < 2; ++layer) {
        const float* xin  = (layer == 0) ? x     : hbuf;
        const float* W    = (layer == 0) ? W0    : W1;
        const float* asrc = (layer == 0) ? asrc0 : asrc1;
        const float* adst = (layer == 0) ? adst0 : adst1;
        const float* bias = (layer == 0) ? b0    : b1;

        gat_gemm_wmma<<<gGemm, 32, 0, stream>>>(xin, W, xw, N);
        gat_alpha<<<gAlpha, 256, 0, stream>>>(xw, asrc, adst, as_, ad_, N);

        fill_f32<<<gNH, 256, 0, stream>>>(mmax, -__builtin_inff(), nNH);
        fill_f32<<<gNH, 256, 0, stream>>>(den, 0.f, nNH);
        fill_f32<<<gNF, 256, 0, stream>>>(acc, 0.f, nNF);

        gat_edge_emax<<<gEH, 256, 0, stream>>>(ei, E, Etot, as_, ad_, ebuf, mmax);
        gat_edge_exp<<<gEH, 256, 0, stream>>>(ei, E, Etot, mmax, ebuf, den);
        gat_scatter<<<gScat, 256, 0, stream>>>(ei, E, Etot, xw, ebuf, den, acc);
        gat_bias_elu<<<gNF, 256, 0, stream>>>(acc, bias, hbuf, nNF);
    }

    fill_f32<<<(G * FDIM + G + 255) / 256, 256, 0, stream>>>(pooled, 0.f,
                                                             (long)G * FDIM + G);
    pool_accum<<<gPool, 256, 0, stream>>>(hbuf, batch, pooled, cnt, N);
    final_linear<<<G, 32, 0, stream>>>(pooled, cnt, lin_w, lin_b, outp);
}
